// GATLayer_30889404793458
// MI455X (gfx1250) — compile-verified, matrix operand order checked by reading, not run on hardware
//
#include <hip/hip_runtime.h>
#include <hip/hip_bf16.h>

// Problem constants (match reference)
constexpr int kN   = 50000;
constexpr int kE   = 800000;
constexpr int kDIN = 256;
constexpr int kH   = 4;
constexpr int kD   = 64;          // per-head out dim
constexpr int kHD  = kH * kD;     // 256
constexpr float kNegSlope = 0.2f;

typedef __attribute__((ext_vector_type(16))) _Float16 v16h;
typedef __attribute__((ext_vector_type(8)))  _Float16 v8h;
typedef __attribute__((ext_vector_type(4)))  _Float16 v4h;
typedef __attribute__((ext_vector_type(8)))  float    v8f;

// ---------------------------------------------------------------------------
// float atomic max via int/uint punning (m initialized to -inf)
// ---------------------------------------------------------------------------
__device__ __forceinline__ void atomicMaxF(float* addr, float val) {
    if (val >= 0.0f)
        atomicMax((int*)addr, __float_as_int(val));
    else
        atomicMin((unsigned int*)addr, __float_as_uint(val));
}

// ---------------------------------------------------------------------------
// K0: per-launch init: zero accumulator (d_out), m = -inf, s = 0
// ---------------------------------------------------------------------------
__global__ void k_init(float* __restrict__ out, float* __restrict__ m,
                       float* __restrict__ s) {
    int i = blockIdx.x * blockDim.x + threadIdx.x;
    if (i < kN * kHD) out[i] = 0.0f;
    if (i < kN * kH) {
        m[i] = -__builtin_inff();
        s[i] = 0.0f;
    }
}

// ---------------------------------------------------------------------------
// K1a: pack W (f32, [256 x 256] row-major) into f16 B-fragment order.
// Fragment index: wp[((cb*8 + ks)*32 + lane)*16 + j]
//   col = cb*16 + (lane&15)
//   k   = ks*32 + (j&7) + (j>=8 ? 16 : 0) + ((lane>=16) ? 8 : 0)
// so each lane's B fragment for (cb, ks) is one contiguous 32-byte load.
// ---------------------------------------------------------------------------
__global__ void k_pack_w(const float* __restrict__ W, _Float16* __restrict__ wp) {
    int i = blockIdx.x * blockDim.x + threadIdx.x;   // 0 .. 65535
    if (i >= kDIN * kHD) return;
    const int j    = i & 15;
    const int lane = (i >> 4) & 31;
    const int ks   = (i >> 9) & 7;
    const int cb   = i >> 12;
    const int k    = ks * 32 + (j & 7) + ((j >> 3) << 4) + ((lane >> 4) << 3);
    const int col  = cb * 16 + (lane & 15);
    wp[i] = (_Float16)W[(long)k * kHD + col];
}

// ---------------------------------------------------------------------------
// K1b: convert x to f16 row-major (A fragments become contiguous b128 loads)
// ---------------------------------------------------------------------------
__global__ void k_x_to_f16(const float* __restrict__ x, _Float16* __restrict__ xh) {
    long i = (long)blockIdx.x * blockDim.x + threadIdx.x; // float4 index
    if (i >= (long)kN * kHD / 4) return;
    const float4 v = ((const float4*)x)[i];
    v4h o;
    o[0] = (_Float16)v.x; o[1] = (_Float16)v.y;
    o[2] = (_Float16)v.z; o[3] = (_Float16)v.w;
    ((v4h*)xh)[i] = o;
}

// ---------------------------------------------------------------------------
// K1c: h = x @ W via WMMA f32_16x16x32_f16.
// Block = 256 threads (8 waves) owns one 16-row block; each wave computes TWO
// 16x16 column tiles (cb0 = wave, cb1 = wave+8), sharing the A fragment.
// K loop fully unrolled: 16 v_wmma per wave.
// A (16x32 f16): m = lane&15, half j -> k = (j&7)+(j>=8?16:0)+((lane>=16)?8:0)
//   -> per lane: two contiguous 8-half chunks of the f16 row.
// C (16x16 f32): vgpr r -> row = r + ((lane>=16)?8:0), col = lane&15.
// ---------------------------------------------------------------------------
__global__ __launch_bounds__(256) void k_gemm_wmma(
    const _Float16* __restrict__ xh, const _Float16* __restrict__ wp,
    float* __restrict__ h) {
    const int lane = threadIdx.x & 31;
    const int wave = threadIdx.x >> 5;      // 0..7
    const int i0   = blockIdx.x * 16;       // row block
    const int l    = lane & 15;
    const int hi   = (lane >> 4) & 1;

    const _Float16* __restrict__ xrow = xh + (long)(i0 + l) * kDIN;
    const int cb0 = wave, cb1 = wave + 8;
    const _Float16* __restrict__ wp0 = wp + ((long)cb0 * 8 * 32 + lane) * 16;
    const _Float16* __restrict__ wp1 = wp + ((long)cb1 * 8 * 32 + lane) * 16;

    v8f c0 = {}, c1 = {};
#pragma unroll
    for (int ks = 0; ks < 8; ++ks) {
        const int kbase = ks * 32 + (hi << 3);
        const v8h a0 = *(const v8h*)(xrow + kbase);        // k: base+0..7
        const v8h a1 = *(const v8h*)(xrow + kbase + 16);   // k: base+16..23
        const v16h a = __builtin_shufflevector(a0, a1,
            0, 1, 2, 3, 4, 5, 6, 7, 8, 9, 10, 11, 12, 13, 14, 15);
        const v16h b0 = *(const v16h*)(wp0 + ks * 512);    // 32 lanes * 16 halves
        const v16h b1 = *(const v16h*)(wp1 + ks * 512);
        c0 = __builtin_amdgcn_wmma_f32_16x16x32_f16(false, a, false, b0,
                                                    (short)0, c0, false, false);
        c1 = __builtin_amdgcn_wmma_f32_16x16x32_f16(false, a, false, b1,
                                                    (short)0, c1, false, false);
    }
#pragma unroll
    for (int r = 0; r < 8; ++r) {
        const int row = i0 + r + (hi << 3);
        h[(long)row * kHD + cb0 * 16 + l] = c0[r];
        h[(long)row * kHD + cb1 * 16 + l] = c1[r];
    }
}

// ---------------------------------------------------------------------------
// K2: a_src[n,hh] = dot(h[n,hh,:], att_src[hh,:]);  same for a_dst
// ---------------------------------------------------------------------------
__global__ void k_attn_dots(const float* __restrict__ h,
                            const float* __restrict__ att_src,
                            const float* __restrict__ att_dst,
                            float* __restrict__ asrc, float* __restrict__ adst) {
    int i = blockIdx.x * blockDim.x + threadIdx.x;   // n*kH + hh
    if (i >= kN * kH) return;
    const int n = i >> 2, hh = i & 3;
    const float4* hv = (const float4*)(h + (long)n * kHD + hh * kD);
    const float4* as = (const float4*)(att_src + hh * kD);
    const float4* ad = (const float4*)(att_dst + hh * kD);
    float s1 = 0.0f, s2 = 0.0f;
#pragma unroll
    for (int q = 0; q < kD / 4; ++q) {
        const float4 hx = hv[q];
        const float4 a1 = as[q];
        const float4 a2 = ad[q];
        s1 += hx.x * a1.x + hx.y * a1.y + hx.z * a1.z + hx.w * a1.w;
        s2 += hx.x * a2.x + hx.y * a2.y + hx.z * a2.z + hx.w * a2.w;
    }
    asrc[i] = s1;
    adst[i] = s2;
}

// ---------------------------------------------------------------------------
// K3: edge logits e = leaky_relu(a_src[src]+a_dst[dst]); segment max into m
// ---------------------------------------------------------------------------
__global__ void k_edge_logits(const int* __restrict__ src,
                              const int* __restrict__ dst,
                              const float* __restrict__ asrc,
                              const float* __restrict__ adst,
                              float* __restrict__ ebuf, float* __restrict__ m) {
    int i = blockIdx.x * blockDim.x + threadIdx.x;   // e*kH + hh
    if (i >= kE * kH) return;
    const int e = i >> 2, hh = i & 3;
    const int sn = src[e], dn = dst[e];
    float v = asrc[sn * kH + hh] + adst[dn * kH + hh];
    v = v > 0.0f ? v : kNegSlope * v;
    ebuf[i] = v;
    atomicMaxF(&m[dn * kH + hh], v);
}

// ---------------------------------------------------------------------------
// K4: ex = exp(e - m[dst]); segment sum into s; ebuf overwritten with ex
// ---------------------------------------------------------------------------
__global__ void k_edge_exp(const int* __restrict__ dst,
                           const float* __restrict__ m,
                           float* __restrict__ ebuf, float* __restrict__ s) {
    int i = blockIdx.x * blockDim.x + threadIdx.x;   // e*kH + hh
    if (i >= kE * kH) return;
    const int e = i >> 2, hh = i & 3;
    const int dn = dst[e];
    const float ex = __expf(ebuf[i] - m[dn * kH + hh]);
    ebuf[i] = ex;
    atomicAdd(&s[dn * kH + hh], ex);
}

// ---------------------------------------------------------------------------
// K5: out[dst, hh, :] += (ex / s[dst]) * h[src, hh, :]
// one thread per (edge, head, float4 quad): 4 atomic f32 adds each.
// h and out are both L2-resident (51 MB each vs 192 MB L2).
// ---------------------------------------------------------------------------
__global__ void k_aggregate(const int* __restrict__ src,
                            const int* __restrict__ dst,
                            const float* __restrict__ h,
                            const float* __restrict__ ebuf,
                            const float* __restrict__ s,
                            float* __restrict__ out) {
    long i = (long)blockIdx.x * blockDim.x + threadIdx.x; // e*64 + hh*16 + q
    if (i >= (long)kE * kH * (kD / 4)) return;
    const int e  = (int)(i >> 6);
    const int hh = ((int)i >> 4) & 3;
    const int q  = (int)i & 15;
    const int sn = src[e], dn = dst[e];
    const float sv = s[dn * kH + hh];
    const float alpha = ebuf[e * kH + hh] / (sv > 0.0f ? sv : 1.0f);
    const float4 hv = *(const float4*)(h + (long)sn * kHD + hh * kD + q * 4);
    float* o = out + (long)dn * kHD + hh * kD + q * 4;
    atomicAdd(o + 0, alpha * hv.x);
    atomicAdd(o + 1, alpha * hv.y);
    atomicAdd(o + 2, alpha * hv.z);
    atomicAdd(o + 3, alpha * hv.w);
}

// ---------------------------------------------------------------------------
// K6: out = sigmoid(out + bias), in place
// ---------------------------------------------------------------------------
__global__ void k_finalize(float* __restrict__ out,
                           const float* __restrict__ bias) {
    int i = blockIdx.x * blockDim.x + threadIdx.x;
    if (i >= kN * kHD) return;
    const float v = out[i] + bias[i & (kHD - 1)];
    out[i] = 1.0f / (1.0f + __expf(-v));
}

// ---------------------------------------------------------------------------
// Launch: inputs in setup_inputs() order:
//   0: x [N,256] f32      1: edge_index [2,E] int   2: edge_weight [E] f32 (ignored)
//   3: W [256,256] f32    4: att_src [H,64] f32     5: att_dst [H,64] f32
//   6: bias [256] f32
// ---------------------------------------------------------------------------
extern "C" void kernel_launch(void* const* d_in, const int* in_sizes, int n_in,
                              void* d_out, int out_size, void* d_ws, size_t ws_size,
                              hipStream_t stream) {
    const float* x       = (const float*)d_in[0];
    const int*   ei      = (const int*)d_in[1];
    const float* W       = (const float*)d_in[3];
    const float* att_src = (const float*)d_in[4];
    const float* att_dst = (const float*)d_in[5];
    const float* bias    = (const float*)d_in[6];
    float* out = (float*)d_out;

    const int* src = ei;        // edge_index[0]
    const int* dst = ei + kE;   // edge_index[1]

    // Workspace layout
    float* ws   = (float*)d_ws;
    float* h    = ws;                      // N*256 f32 = 51.2 MB
    float* asrc = h    + (long)kN * kHD;   // N*H
    float* adst = asrc + (long)kN * kH;
    float* m    = adst + (long)kN * kH;
    float* s    = m    + (long)kN * kH;
    float* ebuf = s    + (long)kN * kH;    // E*H f32 = 12.8 MB
    _Float16* xh    = (_Float16*)(ebuf + (long)kE * kH);  // N*256 f16 = 25.6 MB
    _Float16* wpack = xh + (long)kN * kHD;                // 64K f16 = 128 KB

    // K0: init accumulator / m / s
    k_init<<<(kN * kHD + 255) / 256, 256, 0, stream>>>(out, m, s);
    // K1a: pack W into f16 B-fragment layout (one-time per launch, 256 KB read)
    k_pack_w<<<(kDIN * kHD + 255) / 256, 256, 0, stream>>>(W, wpack);
    // K1b: x -> f16 row-major
    k_x_to_f16<<<(kN * kHD / 4 + 255) / 256, 256, 0, stream>>>(x, xh);
    // K1c: WMMA GEMM h = x @ W (3125 row blocks, 8 waves x 2 col tiles each)
    k_gemm_wmma<<<kN / 16, 256, 0, stream>>>(xh, wpack, h);
    // K2: attention dots
    k_attn_dots<<<(kN * kH + 255) / 256, 256, 0, stream>>>(h, att_src, att_dst,
                                                           asrc, adst);
    // K3: edge logits + segment max
    k_edge_logits<<<(kE * kH + 255) / 256, 256, 0, stream>>>(src, dst, asrc,
                                                             adst, ebuf, m);
    // K4: exp + segment sum
    k_edge_exp<<<(kE * kH + 255) / 256, 256, 0, stream>>>(dst, m, ebuf, s);
    // K5: weighted scatter aggregation
    {
        long total = (long)kE * kH * (kD / 4);           // 51.2M threads
        k_aggregate<<<(int)((total + 255) / 256), 256, 0, stream>>>(src, dst, h,
                                                                    ebuf, s, out);
    }
    // K6: bias + sigmoid
    k_finalize<<<(kN * kHD + 255) / 256, 256, 0, stream>>>(out, bias);
}
